// SupConDualProbe_33466385170707
// MI455X (gfx1250) — matrix operand order repaired and checked by reading, not compiled, not run on hardware
//
#include <hip/hip_runtime.h>
#include <hip/hip_bf16.h>

typedef __attribute__((ext_vector_type(16))) __bf16 v16bf;
typedef __attribute__((ext_vector_type(8)))  float  v8f;

#define B_ROWS 32768
#define FEAT   4096
#define XDIM   4097
#define HID    2048
#define PROJ   512

// round-to-nearest-even f32 -> bf16
__device__ __forceinline__ unsigned short f2bf(float f) {
  union { float f; unsigned u; } v; v.f = f;
  unsigned r = v.u + 0x7FFFu + ((v.u >> 16) & 1u);
  return (unsigned short)(r >> 16);
}

union FragU { uint4 q[2]; v16bf v; };

// A fragment (16x32 bf16): lanes 0-15 hold K=0..7,16..23; lanes 16-31 hold K=8..15,24..31.
// LDS tile rows are K-contiguous with 80-byte row stride (16B aligned, conflict-padded).
__device__ __forceinline__ v16bf ldsA_frag(const char* base, int row, int kh) {
  FragU u;
  const char* p = base + row * 80 + kh * 16;
  u.q[0] = *(const uint4*)(p);
  u.q[1] = *(const uint4*)(p + 32);
  return u.v;
}
// B fragment (32x16 bf16): lanes 0-15 hold K=0..7,16..23; lanes 16-31 hold K=8..15,24..31.
__device__ __forceinline__ v16bf ldsB_frag(const char* base, int row, int kh) {
  FragU u;
  const char* p = base + row * 80 + kh * 16;
  u.q[0] = *(const uint4*)(p);
  u.q[1] = *(const uint4*)(p + 32);
  return u.v;
}

// CDNA5 async copy: memory -> LDS, no data VGPRs, tracked by ASYNCcnt.
// GVS mode: addr = SGPR64 base + VGPR32 byte offset. ldsaddr = wave-relative
// LDS byte offset (low 32 bits of the flat shared address).
__device__ __forceinline__ void async_g2l_b128(unsigned ldsaddr, unsigned goff,
                                               const void* base) {
  asm volatile("global_load_async_to_lds_b128 %0, %1, %2 offset:0"
               :: "v"(ldsaddr), "v"(goff), "s"(base) : "memory");
}
__device__ __forceinline__ void wait_async0() {
  asm volatile("s_wait_asynccnt 0x0" ::: "memory");
}
// async loads complete in order; <=4 outstanding == oldest tile (4 loads) landed
__device__ __forceinline__ void wait_async4() {
  asm volatile("s_wait_asynccnt 0x4" ::: "memory");
}
__device__ __forceinline__ unsigned lds_off(const void* p) {
  return (unsigned)(size_t)p;   // low 32 bits of flat shared addr == LDS offset
}

// ---------------------------------------------------------------------------
// Prepass: per-row flag partition (nat rows packed from slot 0 upward, oth
// rows from slot B-1 downward) + fp32 -> bf16 conversion of the 4096 features.
// ---------------------------------------------------------------------------
__global__ __launch_bounds__(256) void k_partition(const float* __restrict__ x,
    unsigned short* __restrict__ Xb, int* __restrict__ slot2row,
    int* __restrict__ counters) {
  const int row = blockIdx.x;
  const float* xr = x + (size_t)row * XDIM;
  __shared__ int s_slot;
  if (threadIdx.x == 0) {
    int slot;
    if (xr[FEAT] > 0.5f) slot = atomicAdd(&counters[0], 1);
    else                 slot = (B_ROWS - 1) - atomicAdd(&counters[1], 1);
    slot2row[slot] = row;
    s_slot = slot;
  }
  __syncthreads();
  const int slot = s_slot;
  const int base = threadIdx.x * 16;
  unsigned p[8];
#pragma unroll
  for (int i = 0; i < 8; i++) {
    float f0 = xr[base + 2 * i];
    float f1 = xr[base + 2 * i + 1];
    p[i] = (unsigned)f2bf(f0) | ((unsigned)f2bf(f1) << 16);
  }
  uint4* dst = (uint4*)(Xb + (size_t)slot * FEAT + base);
  dst[0] = make_uint4(p[0], p[1], p[2], p[3]);
  dst[1] = make_uint4(p[4], p[5], p[6], p[7]);
}

// ---------------------------------------------------------------------------
// Weight prep: W (K x N f32, row-major) -> WT (N x K bf16) so each output
// column's K values are contiguous (two b128 LDS reads per B-fragment lane).
// ---------------------------------------------------------------------------
__global__ __launch_bounds__(256) void k_transposeW(const float* __restrict__ W,
    unsigned short* __restrict__ WT, int K, int N) {
  __shared__ float t[32][33];
  const int tx = threadIdx.x & 31, ty = threadIdx.x >> 5;
  const int n0 = blockIdx.x * 32, k0 = blockIdx.y * 32;
#pragma unroll
  for (int r = 0; r < 4; r++)
    t[ty + r * 8][tx] = W[(size_t)(k0 + ty + r * 8) * N + n0 + tx];
  __syncthreads();
#pragma unroll
  for (int r = 0; r < 4; r++) {
    int n = ty + r * 8;
    WT[(size_t)(n0 + n) * K + k0 + tx] = f2bf(t[tx][n]);
  }
}

// ---------------------------------------------------------------------------
// GEMM1: H[slot] = relu(Xb[slot] @ W1(path) + b1(path)), bf16 out.
// Block tile 128x128, K=4096, 8 waves (2x4). Triple-buffered LDS filled by
// GLOBAL_LOAD_ASYNC_TO_LDS_B128, 2-tile-deep async pipeline (wait <= 4).
// ---------------------------------------------------------------------------
__global__ __launch_bounds__(256) void k_gemm1(const unsigned short* __restrict__ Xb,
    const unsigned short* __restrict__ W1T, const float* __restrict__ b1n,
    const float* __restrict__ b1o, unsigned short* __restrict__ Hb,
    const int* __restrict__ counters) {
  const int n_nat = counters[0];
  const int rowBlk = blockIdx.x, colBlk = blockIdx.y, path = blockIdx.z;
  if (path == 0) { if (rowBlk * 128 >= n_nat) return; }
  else           { if (rowBlk * 128 + 128 <= n_nat) return; }

  __shared__ char As[3][10240];   // 128 rows x 32 K bf16, 80B row stride
  __shared__ char Bs[3][10240];

  const int tid = threadIdx.x;
  const int lane = tid & 31, wave = tid >> 5;
  const int waveM = wave & 1, waveN = wave >> 1;
  const int lane15 = lane & 15, kh = lane >> 4;

  const unsigned short* Ag = Xb + (size_t)rowBlk * 128 * FEAT;
  const unsigned short* Bg = W1T + ((size_t)path * HID + (size_t)colBlk * 128) * FEAT;
  const float* biasp = path ? b1o : b1n;

  const unsigned ldsA0 = lds_off(&As[0][0]);
  const unsigned ldsB0 = lds_off(&Bs[0][0]);
  // per-thread chunk coordinates (2 x 16B chunks of each 128x32 tile)
  const int e0 = tid, e1 = tid + 256;
  const unsigned goA0 = (unsigned)((e0 >> 2) * (FEAT * 2) + (e0 & 3) * 16);
  const unsigned goA1 = (unsigned)((e1 >> 2) * (FEAT * 2) + (e1 & 3) * 16);
  const unsigned lo0  = (unsigned)((e0 >> 2) * 80 + (e0 & 3) * 16);
  const unsigned lo1  = (unsigned)((e1 >> 2) * 80 + (e1 & 3) * 16);

  v8f acc[4][2];
#pragma unroll
  for (int i = 0; i < 4; i++)
#pragma unroll
    for (int j = 0; j < 2; j++)
#pragma unroll
      for (int k = 0; k < 8; k++) acc[i][j][k] = 0.f;

  auto issue_tile = [&](int buf, int kbytes) {
    const unsigned d = (unsigned)(buf * 10240);
    async_g2l_b128(ldsA0 + d + lo0, goA0 + kbytes, Ag);
    async_g2l_b128(ldsB0 + d + lo0, goA0 + kbytes, Bg);
    async_g2l_b128(ldsA0 + d + lo1, goA1 + kbytes, Ag);
    async_g2l_b128(ldsB0 + d + lo1, goA1 + kbytes, Bg);
  };
  // B-frags first, then per-mf A-frag immediately consumed: peak frag
  // liveness 24 VGPRs (vs 48) so the accumulators stay pinned.
  auto compute = [&](int buf) {
    v16bf bf0 = ldsB_frag(Bs[buf], waveN * 32 + lane15, kh);
    v16bf bf1 = ldsB_frag(Bs[buf], waveN * 32 + 16 + lane15, kh);
#pragma unroll
    for (int mf = 0; mf < 4; mf++) {
      v16bf a = ldsA_frag(As[buf], waveM * 64 + mf * 16 + lane15, kh);
      acc[mf][0] = __builtin_amdgcn_wmma_f32_16x16x32_bf16(
          false, a, false, bf0, (short)0, acc[mf][0], false, false);
      acc[mf][1] = __builtin_amdgcn_wmma_f32_16x16x32_bf16(
          false, a, false, bf1, (short)0, acc[mf][1], false, false);
    }
  };

  const int NK = FEAT / 32;          // 128 K-tiles, 64 bytes apart
  issue_tile(0, 0);
  issue_tile(1, 64);
  for (int kt = 0; kt < NK; kt++) {
    if (kt + 1 < NK) wait_async4();  // tile kt landed; kt+1 may be in flight
    else             wait_async0();
    __syncthreads();
    compute(kt % 3);
    if (kt + 2 < NK) issue_tile((kt + 2) % 3, (kt + 2) * 64);
  }

  // epilogue: bias + relu + bf16 store, guarded by path's slot range
#pragma unroll
  for (int nf = 0; nf < 2; nf++) {
    const int col = colBlk * 128 + waveN * 32 + nf * 16 + lane15;
    const float bias = biasp[col];
#pragma unroll
    for (int mf = 0; mf < 4; mf++) {
#pragma unroll
      for (int j = 0; j < 8; j++) {
        int rl = waveM * 64 + mf * 16 + kh * 8 + j;
        int slot = rowBlk * 128 + rl;
        bool ok = (path == 0) ? (slot < n_nat) : (slot >= n_nat);
        if (ok) {
          float v = acc[mf][nf][j] + bias;
          v = v > 0.f ? v : 0.f;
          Hb[(size_t)slot * HID + col] = f2bf(v);
        }
      }
    }
  }
}

// ---------------------------------------------------------------------------
// GEMM2 + fused epilogue: z = H @ W2 + b2 (full N=512 per block of 64 rows),
// logits = relu(z) @ Wc + bc, z_norm = z / max(||z||, 1e-12), scattered to
// the original row via slot2row. Waves 4x2 (16 rows x 256 cols each).
// Tiles filled with async-to-LDS; next tile prefetched during compute.
// ---------------------------------------------------------------------------
__global__ __launch_bounds__(256) void k_gemm2(const unsigned short* __restrict__ Hb,
    const unsigned short* __restrict__ W2T,
    const float* __restrict__ b2n, const float* __restrict__ b2o,
    const float* __restrict__ Wcn, const float* __restrict__ Wco,
    const float* __restrict__ bcn, const float* __restrict__ bco,
    const int* __restrict__ slot2row, const int* __restrict__ counters,
    float* __restrict__ out) {
  const int n_nat = counters[0];
  const int rowBlk = blockIdx.x, path = blockIdx.y;
  const int slot0 = rowBlk * 64;
  if (path == 0) { if (slot0 >= n_nat) return; }
  else           { if (slot0 + 64 <= n_nat) return; }

  __shared__ char As[5120];      // 64 x 32 bf16
  __shared__ char Bs[40960];     // 512 x 32 bf16
  __shared__ float red_sq[64][2];
  __shared__ float red_lz[64][2][3];
  __shared__ float s_inv[64];
  __shared__ int   s_orig[64];
  __shared__ int   s_ok[64];

  const int tid = threadIdx.x;
  const int lane = tid & 31, wave = tid >> 5;
  const int waveM = wave & 3, waveN = wave >> 2;
  const int lane15 = lane & 15, kh = lane >> 4;

  const unsigned short* Ag = Hb + (size_t)slot0 * HID;
  const unsigned short* Bg = W2T + (size_t)path * PROJ * HID;
  const unsigned ldsA0 = lds_off(&As[0]);
  const unsigned ldsB0 = lds_off(&Bs[0]);

  v8f acc[16];
#pragma unroll
  for (int i = 0; i < 16; i++)
#pragma unroll
    for (int k = 0; k < 8; k++) acc[i][k] = 0.f;

  const int NK = HID / 32;
  for (int kt = 0; kt < NK; kt++) {
    {
      int r = tid >> 2, c = tid & 3;   // 256 chunks = 64 rows x 4
      async_g2l_b128(ldsA0 + (unsigned)(r * 80 + c * 16),
                     (unsigned)(r * (HID * 2) + kt * 64 + c * 16), Ag);
    }
#pragma unroll
    for (int i = 0; i < 8; i++) {      // 2048 chunks = 512 rows x 4
      int e = tid + i * 256, r = e >> 2, c = e & 3;
      async_g2l_b128(ldsB0 + (unsigned)(r * 80 + c * 16),
                     (unsigned)(r * (HID * 2) + kt * 64 + c * 16), Bg);
    }
    if (kt + 1 < NK) {                 // warm L2/WGP$ for next tile
      __builtin_prefetch(Ag + (size_t)(tid >> 2) * HID + (kt + 1) * 32 + (tid & 3) * 8, 0, 0);
      __builtin_prefetch(Bg + (size_t)(tid >> 2) * HID + (kt + 1) * 32 + (tid & 3) * 8, 0, 0);
    }
    wait_async0();
    __syncthreads();
    v16bf af = ldsA_frag(As, waveM * 16 + lane15, kh);
#pragma unroll
    for (int nf = 0; nf < 16; nf++) {
      v16bf bfm = ldsB_frag(Bs, waveN * 256 + nf * 16 + lane15, kh);
      acc[nf] = __builtin_amdgcn_wmma_f32_16x16x32_bf16(
          false, af, false, bfm, (short)0, acc[nf], false, false);
    }
    __syncthreads();
  }

  const float* b2p = path ? b2o : b2n;
  const float* Wcp = path ? Wco : Wcn;
  const float* bcp = path ? bco : bcn;

  float sq[8], lz[3][8];
#pragma unroll
  for (int j = 0; j < 8; j++) { sq[j] = 0.f; lz[0][j] = 0.f; lz[1][j] = 0.f; lz[2][j] = 0.f; }

#pragma unroll
  for (int nf = 0; nf < 16; nf++) {
    const int col = waveN * 256 + nf * 16 + lane15;
    const float bias = b2p[col];
    const float wc0 = Wcp[col * 3 + 0];
    const float wc1 = Wcp[col * 3 + 1];
    const float wc2 = Wcp[col * 3 + 2];
#pragma unroll
    for (int j = 0; j < 8; j++) {
      float v = acc[nf][j] + bias;
      acc[nf][j] = v;
      sq[j] += v * v;
      float rv = v > 0.f ? v : 0.f;
      lz[0][j] += rv * wc0;
      lz[1][j] += rv * wc1;
      lz[2][j] += rv * wc2;
    }
  }
  // reduce across the 16 lanes of each half (wave32, halves hold disjoint rows)
#pragma unroll
  for (int m = 1; m < 16; m <<= 1) {
#pragma unroll
    for (int j = 0; j < 8; j++) {
      sq[j]    += __shfl_xor(sq[j], m, 32);
      lz[0][j] += __shfl_xor(lz[0][j], m, 32);
      lz[1][j] += __shfl_xor(lz[1][j], m, 32);
      lz[2][j] += __shfl_xor(lz[2][j], m, 32);
    }
  }
  if (lane15 == 0) {
#pragma unroll
    for (int j = 0; j < 8; j++) {
      int r = waveM * 16 + kh * 8 + j;
      red_sq[r][waveN]    = sq[j];
      red_lz[r][waveN][0] = lz[0][j];
      red_lz[r][waveN][1] = lz[1][j];
      red_lz[r][waveN][2] = lz[2][j];
    }
  }
  __syncthreads();
  if (tid < 64) {
    const int r = tid, slot = slot0 + r;
    const bool ok = (path == 0) ? (slot < n_nat) : (slot >= n_nat);
    const int orig = ok ? slot2row[slot] : 0;
    const float ssq = red_sq[r][0] + red_sq[r][1];
    const float inv = 1.f / fmaxf(sqrtf(ssq), 1e-12f);
    s_inv[r] = inv;
    s_orig[r] = orig;
    s_ok[r] = ok ? 1 : 0;
    if (ok) {
      size_t lo = (size_t)B_ROWS * PROJ + (size_t)orig * 3;
      out[lo + 0] = red_lz[r][0][0] + red_lz[r][1][0] + bcp[0];
      out[lo + 1] = red_lz[r][0][1] + red_lz[r][1][1] + bcp[1];
      out[lo + 2] = red_lz[r][0][2] + red_lz[r][1][2] + bcp[2];
    }
  }
  __syncthreads();
#pragma unroll
  for (int nf = 0; nf < 16; nf++) {
    const int col = waveN * 256 + nf * 16 + lane15;
#pragma unroll
    for (int j = 0; j < 8; j++) {
      int r = waveM * 16 + kh * 8 + j;
      if (s_ok[r])
        out[(size_t)s_orig[r] * PROJ + col] = acc[nf][j] * s_inv[r];
    }
  }
}

// ---------------------------------------------------------------------------
extern "C" void kernel_launch(void* const* d_in, const int* in_sizes, int n_in,
                              void* d_out, int out_size, void* d_ws, size_t ws_size,
                              hipStream_t stream) {
  (void)in_sizes; (void)n_in; (void)out_size; (void)ws_size;
  const float* x   = (const float*)d_in[0];
  const float* W1n = (const float*)d_in[1];
  const float* b1n = (const float*)d_in[2];
  const float* W2n = (const float*)d_in[3];
  const float* b2n = (const float*)d_in[4];
  const float* Wcn = (const float*)d_in[5];
  const float* bcn = (const float*)d_in[6];
  const float* W1o = (const float*)d_in[7];
  const float* b1o = (const float*)d_in[8];
  const float* W2o = (const float*)d_in[9];
  const float* b2o = (const float*)d_in[10];
  const float* Wco = (const float*)d_in[11];
  const float* bco = (const float*)d_in[12];
  float* out = (float*)d_out;

  char* ws = (char*)d_ws;
  size_t off = 0;
  int* counters = (int*)(ws + off);              off += 256;
  int* slot2row = (int*)(ws + off);              off += (size_t)B_ROWS * 4;
  unsigned short* Xb  = (unsigned short*)(ws + off); off += (size_t)B_ROWS * FEAT * 2;
  unsigned short* Hb  = (unsigned short*)(ws + off); off += (size_t)B_ROWS * HID * 2;
  unsigned short* W1T = (unsigned short*)(ws + off); off += (size_t)2 * HID * FEAT * 2;
  unsigned short* W2T = (unsigned short*)(ws + off); off += (size_t)2 * PROJ * HID * 2;

  hipMemsetAsync(counters, 0, 16, stream);
  k_partition<<<B_ROWS, 256, 0, stream>>>(x, Xb, slot2row, counters);
  k_transposeW<<<dim3(HID / 32, FEAT / 32), 256, 0, stream>>>(W1n, W1T, FEAT, HID);
  k_transposeW<<<dim3(HID / 32, FEAT / 32), 256, 0, stream>>>(W1o, W1T + (size_t)HID * FEAT, FEAT, HID);
  k_transposeW<<<dim3(PROJ / 32, HID / 32), 256, 0, stream>>>(W2n, W2T, HID, PROJ);
  k_transposeW<<<dim3(PROJ / 32, HID / 32), 256, 0, stream>>>(W2o, W2T + (size_t)PROJ * HID, HID, PROJ);
  k_gemm1<<<dim3(B_ROWS / 128, HID / 128, 2), 256, 0, stream>>>(Xb, W1T, b1n, b1o, Hb, counters);
  k_gemm2<<<dim3(B_ROWS / 64, 2), 256, 0, stream>>>(Hb, W2T, b2n, b2o, Wcn, Wco, bcn, bco, slot2row, counters, out);
}
